// SpectralAttention_51513837748782
// MI455X (gfx1250) — matrix operand order repaired and checked by reading, not compiled, not run on hardware
//
#include <hip/hip_runtime.h>

// SpectralAttention for MI455X (gfx1250, wave32, WMMA, async-LDS staging).
// All GEMMs via v_wmma_f32_16x16x32_bf16 (bf16 in, f32 accumulate).
// Stage 1: QKV projection (Q,K row-major bf16; V stored transposed [B][C][S]).
// Stage 2: flash-attention; 4 waves/block share double-buffered K/V chunks
//          staged through LDS with GLOBAL_LOAD_ASYNC_TO_LDS_B128 (ASYNCcnt).

#define NB 8
#define NS 2048
#define NC 512

#define KSTR 520   // LDS row stride (bf16 elems) for K chunk: 512 + 8 pad
#define VSTR 40    // LDS row stride (bf16 elems) for V chunk: 32 + 8 pad
#define PSTR 36    // LDS row stride (f32 elems) for P staging: 32 + 4 pad

typedef __attribute__((ext_vector_type(16))) __bf16 v16bf;
typedef __attribute__((ext_vector_type(8)))  __bf16 v8bf;
typedef __attribute__((ext_vector_type(8)))  float  v8f;
typedef __attribute__((ext_vector_type(4)))  float  v4f;

__device__ __forceinline__ v8f zero8() {
    v8f z;
#pragma unroll
    for (int i = 0; i < 8; ++i) z[i] = 0.0f;
    return z;
}

__device__ __forceinline__ v8f wmma_bf16(v16bf a, v16bf b, v8f c) {
    // D = A(16x32) * B(32x16) + C, f32 accumulate
    return __builtin_amdgcn_wmma_f32_16x16x32_bf16(false, a, false, b,
                                                   (short)0, c, false, false);
}

// ---- CDNA5 async global->LDS (tracked by ASYNCcnt) ------------------------
__device__ __forceinline__ void async_b128(unsigned ldsoff, const void* g) {
    asm volatile("global_load_async_to_lds_b128 %0, %1, off"
                 :: "v"(ldsoff), "v"((unsigned long long)(size_t)g)
                 : "memory");
}
__device__ __forceinline__ void wait_async0() {
    asm volatile("s_wait_asynccnt 0" ::: "memory");
}

// Pack 16 f32 -> 16 bf16 (element order preserved)
__device__ __forceinline__ v16bf cvt16(v4f a, v4f b, v4f c, v4f d) {
    v16bf r;
#pragma unroll
    for (int i = 0; i < 4; ++i) {
        r[i]      = (__bf16)a[i];
        r[4 + i]  = (__bf16)b[i];
        r[8 + i]  = (__bf16)c[i];
        r[12 + i] = (__bf16)d[i];
    }
    return r;
}

// B-operand from f32 row-major: 16 contiguous K values for this lane.
__device__ __forceinline__ v16bf b_from_f32(const float* p) {
    return cvt16(*(const v4f*)p, *(const v4f*)(p + 4),
                 *(const v4f*)(p + 8), *(const v4f*)(p + 12));
}

// A-operand from f32 row-major: elems 0..7 = row[clo..+7], 8..15 = row[chi..+7].
__device__ __forceinline__ v16bf a_from_f32(const float* row, int clo, int chi) {
    return cvt16(*(const v4f*)(row + clo), *(const v4f*)(row + clo + 4),
                 *(const v4f*)(row + chi), *(const v4f*)(row + chi + 4));
}

__device__ __forceinline__ v16bf cat8(v8bf a, v8bf b) {
    return __builtin_shufflevector(a, b, 0, 1, 2, 3, 4, 5, 6, 7,
                                   8, 9, 10, 11, 12, 13, 14, 15);
}

// B-operand from bf16 row-major: 16 contiguous K values.
__device__ __forceinline__ v16bf b_from_bf16(const __bf16* p) {
    return cat8(*(const v8bf*)p, *(const v8bf*)(p + 8));
}

// A-operand from bf16 row-major: chunks at clo and clo+8.
__device__ __forceinline__ v16bf a_from_bf16(const __bf16* row, int clo) {
    return cat8(*(const v8bf*)(row + clo), *(const v8bf*)(row + clo + 8));
}

__device__ __forceinline__ float rmax16(float v) {
    v = fmaxf(v, __shfl_xor(v, 1, 32));
    v = fmaxf(v, __shfl_xor(v, 2, 32));
    v = fmaxf(v, __shfl_xor(v, 4, 32));
    v = fmaxf(v, __shfl_xor(v, 8, 32));
    return v;
}

__device__ __forceinline__ float rsum16(float v) {
    v += __shfl_xor(v, 1, 32);
    v += __shfl_xor(v, 2, 32);
    v += __shfl_xor(v, 4, 32);
    v += __shfl_xor(v, 8, 32);
    return v;
}

// ---------------------------------------------------------------------------
// Stage 1: projection. y = x @ W^T + b for one 16-row s-tile, all 512 d.
// vmode 0 (Q,K): swapped operands A=W, B=x^T  -> D tile M=d,N=s; per-lane
//               8 contiguous d -> coalesced b128 store into row-major [s][d].
// vmode 1 (V):  A=x, B=W^T -> D tile M=s,N=d; per-lane 8 contiguous s ->
//               coalesced b128 store into transposed Vt [B][C][S].
// Block = 256 threads = 8 waves; wave w handles d-tiles {w, w+8, w+16, w+24}.
// ---------------------------------------------------------------------------
__global__ __launch_bounds__(256) void qkv_kernel(const float* __restrict__ x,
                                                  const float* __restrict__ W,
                                                  const float* __restrict__ bias,
                                                  __bf16* __restrict__ out,
                                                  const int vmode) {
    const int lane  = threadIdx.x & 31;
    const int wv    = threadIdx.x >> 5;
    const int lm    = lane & 15;
    const int hi    = lane >> 4;           // lane half: 0 or 1
    const int sbase = blockIdx.x * 16;     // row tile into flattened [B*S]

    const float* xr = x + (size_t)(sbase + lm) * NC;

    v16bf xop[16];                         // x operand cached over K=512
    if (vmode == 0) {
#pragma unroll
        for (int cc = 0; cc < 16; ++cc)
            xop[cc] = b_from_f32(xr + cc * 32 + hi * 16);
    } else {
#pragma unroll
        for (int cc = 0; cc < 16; ++cc)
            xop[cc] = a_from_f32(xr, cc * 32 + hi * 16, cc * 32 + hi * 16 + 8);
    }

    for (int i = 0; i < 4; ++i) {
        const int dbase = (wv + i * 8) * 16;
        const float* wr = W + (size_t)(dbase + lm) * NC;
        v8f acc = zero8();

        if (vmode == 0) {
#pragma unroll
            for (int cc = 0; cc < 16; ++cc)
                acc = wmma_bf16(a_from_f32(wr, cc * 32 + hi * 16,
                                           cc * 32 + hi * 16 + 8),
                                xop[cc], acc);
            const float* bp = bias + dbase + hi * 8;
            v8bf o;
#pragma unroll
            for (int r = 0; r < 8; ++r) o[r] = (__bf16)(acc[r] + bp[r]);
            *(v8bf*)(out + (size_t)(sbase + lm) * NC + dbase + hi * 8) = o;
        } else {
#pragma unroll
            for (int cc = 0; cc < 16; ++cc)
                acc = wmma_bf16(xop[cc], b_from_f32(wr + cc * 32 + hi * 16), acc);
            const float bb   = bias[dbase + lm];
            const int   bidx = sbase >> 11;        // batch = sbase / 2048
            const int   sloc = sbase & (NS - 1);
            v8bf o;
#pragma unroll
            for (int r = 0; r < 8; ++r) o[r] = (__bf16)(acc[r] + bb);
            *(v8bf*)(out + ((size_t)bidx * NC + dbase + lm) * NS + sloc + hi * 8) = o;
        }
    }
}

// ---------------------------------------------------------------------------
// Async stage of one 32-key chunk into LDS (shared by the 4 waves).
// K chunk: 32 rows x 512 bf16 (row stride KSTR); wave w loads rows [8w, 8w+8).
// V chunk: 512 rows x 32 bf16 (row stride VSTR); wave w loads rows [128w, ..).
// ---------------------------------------------------------------------------
__device__ __forceinline__ void issue_chunk(const __bf16* kbase, const __bf16* vbase,
                                            int k0, __bf16* kdst, __bf16* vdst,
                                            int w, int lane) {
#pragma unroll
    for (int j = 0; j < 16; ++j) {
        const int row = w * 8 + (j >> 1);
        const int off = (j & 1) * 256 + lane * 8;
        async_b128((unsigned)(size_t)(kdst + row * KSTR + off),
                   kbase + (size_t)(k0 + row) * NC + off);
    }
#pragma unroll
    for (int i = 0; i < 16; ++i) {
        const int d   = w * 128 + i * 8 + (lane >> 2);
        const int sub = (lane & 3) * 8;
        async_b128((unsigned)(size_t)(vdst + d * VSTR + sub),
                   vbase + (size_t)d * NS + k0 + sub);
    }
}

// ---------------------------------------------------------------------------
// Stage 2: flash attention. 4 waves/block, each owns one 16-query tile; the
// block shares double-buffered K/V chunks in LDS (async DMA, ASYNCcnt).
// Online softmax over 64 chunks of 32 keys; Q in registers (16 A-operands);
// 256-VGPR f32 output accumulator per wave.
// ---------------------------------------------------------------------------
__global__ __launch_bounds__(128) void attn_kernel(const __bf16* __restrict__ Qb,
                                                   const __bf16* __restrict__ Kb,
                                                   const __bf16* __restrict__ Vt,
                                                   const float* __restrict__ eig,
                                                   float* __restrict__ out) {
    extern __shared__ char smem[];
    __bf16* kb0 = (__bf16*)smem;                       // 32*KSTR
    __bf16* kb1 = kb0 + 32 * KSTR;
    __bf16* vb0 = kb1 + 32 * KSTR;                     // 512*VSTR
    __bf16* vb1 = vb0 + (size_t)NC * VSTR;
    float*  pb  = (float*)(vb1 + (size_t)NC * VSTR);   // 16*PSTR per wave

    const int   lane  = threadIdx.x & 31;
    const int   w     = threadIdx.x >> 5;
    const int   lm    = lane & 15;
    const int   hi    = lane >> 4;
    const int   qt    = blockIdx.x * 4 + w;            // query tile over [B*S/16]
    const int   b     = qt >> 7;                       // 128 q-tiles per batch
    const float scale = 0.04419417382415922f;          // 1/sqrt(512)

    float* pbuf = pb + w * 16 * PSTR;

    // Q A-operands for this tile (lane m = row qt*16+m), 128 VGPRs total.
    v16bf q[16];
    const __bf16* qrow = Qb + (size_t)(qt * 16 + lm) * NC;
#pragma unroll
    for (int cc = 0; cc < 16; ++cc) q[cc] = a_from_bf16(qrow, cc * 32 + hi * 16);

    v8f acc[32];                                       // out accum: 16 q x 512 d
#pragma unroll
    for (int t = 0; t < 32; ++t) acc[t] = zero8();

    float rowm[8], rowl[8];                            // per-row max / denom
#pragma unroll
    for (int r = 0; r < 8; ++r) { rowm[r] = -3.0e38f; rowl[r] = 0.0f; }

    const float*  eb    = eig + (size_t)b * NS;
    const __bf16* kbase = Kb + (size_t)b * NS * NC;
    const __bf16* vbase = Vt + (size_t)b * NC * NS;

    // prefetch chunk 0 into buffer 0
    issue_chunk(kbase, vbase, 0, kb0, vb0, w, lane);

    for (int kc = 0; kc < NS / 32; ++kc) {
        wait_async0();        // this wave's loads for the current buffer done
        __syncthreads();      // everyone's loads done + everyone left old buffer

        if (kc + 1 < NS / 32) {
            __bf16* kn = (kc & 1) ? kb0 : kb1;
            __bf16* vn = (kc & 1) ? vb0 : vb1;
            issue_chunk(kbase, vbase, (kc + 1) * 32, kn, vn, w, lane);
        }
        const __bf16* kcur = (kc & 1) ? kb1 : kb0;
        const __bf16* vcur = (kc & 1) ? vb1 : vb0;

        const int k0 = kc * 32;

        // per-key factor: scale * sigmoid(eig[key]); lane <-> key (N) in C layout
        const float e0 = scale / (1.0f + __expf(-eb[k0 + lm]));
        const float e1 = scale / (1.0f + __expf(-eb[k0 + 16 + lm]));

        // scores: two 16x16 tiles (keys +0..15 and +16..31), K-dim 512, from LDS
        v8f s0 = zero8(), s1 = zero8();
        const __bf16* kr0 = kcur + (size_t)lm * KSTR + hi * 16;
        const __bf16* kr1 = kcur + (size_t)(16 + lm) * KSTR + hi * 16;
#pragma unroll
        for (int cc = 0; cc < 16; ++cc) {
            s0 = wmma_bf16(q[cc], b_from_bf16(kr0 + cc * 32), s0);
            s1 = wmma_bf16(q[cc], b_from_bf16(kr1 + cc * 32), s1);
        }

        // online softmax update (per row r within the lane half)
        float cf[8];
#pragma unroll
        for (int r = 0; r < 8; ++r) {
            const float a0 = s0[r] * e0;
            const float a1 = s1[r] * e1;
            const float t  = rmax16(fmaxf(a0, a1));
            const float mn = fmaxf(rowm[r], t);
            cf[r]   = __expf(rowm[r] - mn);
            rowm[r] = mn;
            const float p0 = __expf(a0 - mn);
            const float p1 = __expf(a1 - mn);
            pbuf[(r + hi * 8) * PSTR + lm]      = p0;
            pbuf[(r + hi * 8) * PSTR + 16 + lm] = p1;
            rowl[r] = rowl[r] * cf[r] + rsum16(p0 + p1);
        }

        // rescale running output accumulators
#pragma unroll
        for (int t = 0; t < 32; ++t)
#pragma unroll
            for (int r = 0; r < 8; ++r) acc[t][r] *= cf[r];

        // P: C-layout (per-wave LDS, same-wave DS in-order) -> A-operand
        const float* pr = pbuf + lm * PSTR + hi * 16;
        const v16bf  pa = cvt16(*(const v4f*)pr, *(const v4f*)(pr + 4),
                                *(const v4f*)(pr + 8), *(const v4f*)(pr + 12));

        // out += P(16x32) @ V(32x512): B-operand rows contiguous in LDS V chunk
#pragma unroll
        for (int t = 0; t < 32; ++t) {
            const __bf16* vr = vcur + (size_t)(t * 16 + lm) * VSTR + hi * 16;
            acc[t] = wmma_bf16(pa, b_from_bf16(vr), acc[t]);
        }
    }

    // epilogue: divide by softmax denominator, store f32 output
    float inv[8];
#pragma unroll
    for (int r = 0; r < 8; ++r) inv[r] = 1.0f / rowl[r];
#pragma unroll
    for (int t = 0; t < 32; ++t) {
#pragma unroll
        for (int r = 0; r < 8; ++r) {
            out[(size_t)(qt * 16 + hi * 8 + r) * NC + t * 16 + lm] =
                acc[t][r] * inv[r];
        }
    }
}

extern "C" void kernel_launch(void* const* d_in, const int* in_sizes, int n_in,
                              void* d_out, int out_size, void* d_ws, size_t ws_size,
                              hipStream_t stream) {
    const float* x   = (const float*)d_in[0];
    const float* eig = (const float*)d_in[1];
    const float* Wq  = (const float*)d_in[2];
    const float* bq  = (const float*)d_in[3];
    const float* Wk  = (const float*)d_in[4];
    const float* bk  = (const float*)d_in[5];
    const float* Wv  = (const float*)d_in[6];
    const float* bv  = (const float*)d_in[7];
    float*       out = (float*)d_out;

    const size_t n  = (size_t)NB * NS * NC;   // 8M elements per matrix
    __bf16* Qb = (__bf16*)d_ws;               // 16 MB
    __bf16* Kb = Qb + n;                      // 16 MB
    __bf16* Vt = Kb + n;                      // 16 MB (transposed [B][C][S])

    const dim3 g1(NB * NS / 16);              // 1024 s-tiles
    qkv_kernel<<<g1, dim3(256), 0, stream>>>(x, Wq, bq, Qb, 0);
    qkv_kernel<<<g1, dim3(256), 0, stream>>>(x, Wk, bk, Kb, 0);
    qkv_kernel<<<g1, dim3(256), 0, stream>>>(x, Wv, bv, Vt, 1);

    // LDS: 2*K chunk + 2*V chunk + 4 P-staging regions
    const int smem = (int)((2 * 32 * KSTR + 2 * NC * VSTR) * sizeof(__bf16) +
                           4 * 16 * PSTR * sizeof(float));   // 157,696 B
    (void)hipFuncSetAttribute((const void*)attn_kernel,
                              hipFuncAttributeMaxDynamicSharedMemorySize, smem);
    attn_kernel<<<dim3(NB * NS / 16 / 4), dim3(128), smem, stream>>>(Qb, Kb, Vt,
                                                                     eig, out);
}
